// Decoder_51659866637081
// MI455X (gfx1250) — matrix operand order repaired and checked by reading, not compiled, not run on hardware
//
#include <hip/hip_runtime.h>

// Problem dims
#define Bsz  64
#define Tsz  256
#define Hsz  128
#define H4sz 512
#define ROWS 4      // batch rows per decoder workgroup (16 workgroups)

typedef __attribute__((ext_vector_type(8)))  float  v8f;
typedef __attribute__((ext_vector_type(4)))  float  v4f;
typedef __attribute__((ext_vector_type(16))) __bf16 v16bf;
typedef int v4i __attribute__((vector_size(16)));   // matches async-builtin param type

#if __has_builtin(__builtin_amdgcn_global_load_async_to_lds_b128) && \
    __has_builtin(__builtin_amdgcn_s_wait_asynccnt)
#define HAVE_ASYNC_LDS 1
typedef __attribute__((address_space(1))) v4i* gv4i_p;   // global (AS1) v4i*
typedef __attribute__((address_space(3))) v4i* lv4i_p;   // LDS (AS3) v4i*
#endif

static __device__ __forceinline__ float fast_exp(float x) {
  return __builtin_amdgcn_exp2f(x * 1.4426950408889634f);
}
static __device__ __forceinline__ float fast_tanh(float x) {
#if __has_builtin(__builtin_amdgcn_tanhf)
  return __builtin_amdgcn_tanhf(x);          // gfx1250 v_tanh_f32 (trans op)
#else
  x = fminf(fmaxf(x, -15.f), 15.f);
  float t = __builtin_amdgcn_exp2f(x * 2.8853900817779268f);
  return (t - 1.f) * __builtin_amdgcn_rcpf(t + 1.f);
#endif
}
static __device__ __forceinline__ float fast_sigmoid(float x) {
  return 0.5f * fast_tanh(0.5f * x) + 0.5f;  // sigmoid via tanh: one trans op
}

// ---------------------------------------------------------------------------
// Kernel 1: enc_proj[b,t,:] = enc_output[b,t,:] @ W1   (M=16384, N=128, K=128)
// ---------------------------------------------------------------------------
__global__ __launch_bounds__(256) void enc_proj_gemm(const float* __restrict__ enc,
                                                     const float* __restrict__ W1,
                                                     float* __restrict__ ep) {
  const int lane = threadIdx.x & 31;
  const int wid  = threadIdx.x >> 5;
  const int gw   = blockIdx.x * 8 + wid;          // 0..8191 tiles
  const int mt   = gw >> 3, nt = gw & 7;
  const int row0 = mt * 16, n0 = nt * 16;
  const int hl   = lane >> 4, r = lane & 15;

  const float* arow = enc + (size_t)(row0 + r) * Hsz;
  v8f acc = {};
#pragma unroll
  for (int kc = 0; kc < 4; ++kc) {
    v16bf a, bm;
#pragma unroll
    for (int j = 0; j < 16; ++j)   // 16-bit A layout (ISA 7.12.2)
      a[j] = (__bf16)arow[32 * kc + (j & 7) + 8 * hl + ((j & 8) ? 16 : 0)];
    const float* brow = W1 + (size_t)(32 * kc) * Hsz + n0 + r;   // B: lane=N, elem=K (table order)
#pragma unroll
    for (int j = 0; j < 16; ++j) bm[j] = (__bf16)brow[(size_t)((j & 7) + 8 * hl + ((j & 8) ? 16 : 0)) * Hsz];
    acc = __builtin_amdgcn_wmma_f32_16x16x32_bf16(false, a, false, bm, (short)0, acc,
                                                  false, false);
  }
#pragma unroll
  for (int i = 0; i < 8; ++i)      // C/D: VGPR i -> rows i / i+8
    ep[(size_t)(row0 + i + 8 * hl) * Hsz + n0 + r] = acc[i];
}

// ---------------------------------------------------------------------------
// Kernel 2: convert Wr and W2 to bf16 once
// ---------------------------------------------------------------------------
__global__ __launch_bounds__(256) void weights_to_bf16(const float* __restrict__ Wr,
                                                       const float* __restrict__ W2,
                                                       __bf16* __restrict__ wrbf,
                                                       __bf16* __restrict__ w2bf) {
  int idx = blockIdx.x * 256 + threadIdx.x;
  if (idx < Hsz * H4sz) wrbf[idx] = (__bf16)Wr[idx];
  else                  w2bf[idx - Hsz * H4sz] = (__bf16)W2[idx - Hsz * H4sz];
}

// ---------------------------------------------------------------------------
// Kernel 3: persistent decoder. 16 workgroups x 256 threads (8 wave32).
// Workgroup g owns batch rows [4g,4g+4). WMMA tiles stay M=16 with zero
// padding rows (D-row m depends only on A/C-row m, so live rows are exact).
// ---------------------------------------------------------------------------
__global__ __launch_bounds__(256) void ptrnet_decoder(
    const float* __restrict__ x, const float* __restrict__ ep,
    const __bf16* __restrict__ wrbf, const __bf16* __restrict__ w2bf,
    const float* __restrict__ h0, const float* __restrict__ c0,
    const float* __restrict__ Vv, const float* __restrict__ Wk,
    const float* __restrict__ bias, float* __restrict__ out) {
  __shared__ __align__(16) __bf16 sW2[Hsz][Hsz + 8];   // 34.8 KB attention weight
  __shared__ float  sWk[2][H4sz];                      //  4 KB
  __shared__ float  sB[H4sz];                          //  2 KB
  __shared__ float  sV[Hsz];                           //  0.5 KB
  __shared__ float  sC[ROWS][Hsz];                     //  2 KB cell state (f32)
  __shared__ __bf16 sH[16][Hsz + 8];                   //  4.3 KB hidden (bf16 WMMA A)
  __shared__ float  sZ[16][H4sz + 8];                  // 33 KB gate pre-activations
  __shared__ float  sA[16][Hsz + 4];                   //  8.4 KB h @ W2
  __shared__ float  sS[ROWS][Tsz + 4];                 //  4.2 KB scores
  __shared__ float  sP[ROWS][2];                       //  ptr (prev step)

  const int tid  = threadIdx.x;
  const int lane = tid & 31;
  const int wid  = tid >> 5;
  const int hl   = lane >> 4;
  const int r    = lane & 15;
  const int b0   = blockIdx.x * ROWS;

  // ---- one-time init ----
#ifdef HAVE_ASYNC_LDS
  for (int e = tid; e < Hsz * 16; e += 256) {          // 128 rows x 16 chunks of 16B
    int row = e >> 4, ch = e & 15;
    __builtin_amdgcn_global_load_async_to_lds_b128(
        (gv4i_p)((const char*)w2bf + row * 256 + ch * 16),
        (lv4i_p)((char*)&sW2[row][0] + ch * 16),
        0, 0);
  }
#else
  for (int e = tid; e < Hsz * Hsz; e += 256) sW2[e >> 7][e & 127] = w2bf[e];
#endif
  for (int e = tid; e < 2 * H4sz; e += 256) sWk[e >> 9][e & 511] = Wk[e];
  for (int e = tid; e < H4sz; e += 256) sB[e] = bias[e];
  if (tid < Hsz) sV[tid] = Vv[tid];
  for (int e = tid; e < 16 * Hsz; e += 256) sH[e >> 7][e & 127] = (__bf16)0.0f;
  for (int e = tid; e < 16 * H4sz; e += 256) sZ[e >> 9][e & 511] = 0.0f;
  for (int e = tid; e < ROWS * Hsz; e += 256) {
    int bi = e >> 7, k = e & 127;
    sC[bi][k] = c0[(b0 + bi) * Hsz + k];
    sH[bi][k] = (__bf16)h0[(b0 + bi) * Hsz + k];
  }
  if (tid < ROWS * 2) sP[tid >> 1][tid & 1] = 1.0f;    // ptr0 = ones
#ifdef HAVE_ASYNC_LDS
  __builtin_amdgcn_s_wait_asynccnt(0);
#endif
  __syncthreads();

  for (int step = 0; step < Tsz; ++step) {
    // (1) z = bias + ptr @ Wk           [ROWS,512]
    for (int e = tid; e < ROWS * H4sz; e += 256) {
      int bi = e >> 9, j = e & 511;
      sZ[bi][j] = sB[j] + sP[bi][0] * sWk[0][j] + sP[bi][1] * sWk[1][j];
    }
    __syncthreads();

    // (2) z += h_prev @ Wr   (WMMA: M=16, N=512 -> 4 n-tiles/wave, K=128)
#pragma unroll
    for (int it = 0; it < 4; ++it) {
      const int n0 = (wid * 4 + it) * 16;
      v8f acc;
#pragma unroll
      for (int i = 0; i < 8; ++i) acc[i] = sZ[i + 8 * hl][n0 + r];
#pragma unroll
      for (int kc = 0; kc < 4; ++kc) {
        v16bf a, bm;
#pragma unroll
        for (int j = 0; j < 16; ++j)
          a[j] = sH[r][32 * kc + (j & 7) + 8 * hl + ((j & 8) ? 16 : 0)];
        const __bf16* brow = wrbf + (size_t)(32 * kc) * H4sz + n0 + r;
#pragma unroll
        for (int j = 0; j < 16; ++j) bm[j] = brow[(size_t)((j & 7) + 8 * hl + ((j & 8) ? 16 : 0)) * H4sz];
        acc = __builtin_amdgcn_wmma_f32_16x16x32_bf16(false, a, false, bm, (short)0, acc,
                                                      false, false);
      }
#pragma unroll
      for (int i = 0; i < 8; ++i) sZ[i + 8 * hl][n0 + r] = acc[i];
    }
    __syncthreads();

    // (3) LSTM gates (order i,f,g,o) -> c_new, h_new
    for (int e = tid; e < ROWS * Hsz; e += 256) {
      int bi = e >> 7, k = e & 127;
      float ig = fast_sigmoid(sZ[bi][k]);
      float fg = fast_sigmoid(sZ[bi][Hsz + k]);
      float gg = fast_tanh(sZ[bi][2 * Hsz + k]);
      float og = fast_sigmoid(sZ[bi][3 * Hsz + k]);
      float cn = fg * sC[bi][k] + ig * gg;
      sC[bi][k] = cn;
      sH[bi][k] = (__bf16)(og * fast_tanh(cn));
    }
    __syncthreads();

    // (4) sA = h_new @ W2   (WMMA: M=16, N=128 -> 1 n-tile/wave, K=128)
    {
      const int n0 = wid * 16;
      v8f acc = {};
#pragma unroll
      for (int kc = 0; kc < 4; ++kc) {
        v16bf a, bm;
#pragma unroll
        for (int j = 0; j < 16; ++j)
          a[j] = sH[r][32 * kc + (j & 7) + 8 * hl + ((j & 8) ? 16 : 0)];
#pragma unroll
        for (int j = 0; j < 16; ++j) bm[j] = sW2[32 * kc + (j & 7) + 8 * hl + ((j & 8) ? 16 : 0)][n0 + r];
        acc = __builtin_amdgcn_wmma_f32_16x16x32_bf16(false, a, false, bm, (short)0, acc,
                                                      false, false);
      }
#pragma unroll
      for (int i = 0; i < 8; ++i) sA[i + 8 * hl][n0 + r] = acc[i];
    }
    __syncthreads();

    // (5) scores[bi][t] = sum_k tanh(enc_proj + hW2) * V  (enc_proj L2-resident)
    for (int p = tid; p < ROWS * Tsz; p += 256) {
      int bi = p & (ROWS - 1), t = p >> 2;
      const float* erow = ep + ((size_t)(b0 + bi) * Tsz + t) * Hsz;
      if (t + 64 < Tsz) __builtin_prefetch(erow + 64 * Hsz, 0, 0);
      float s = 0.f;
      for (int k = 0; k < Hsz; k += 4) {
        v4f e4 = *(const v4f*)(erow + k);
        s += fast_tanh(e4.x + sA[bi][k + 0]) * sV[k + 0];
        s += fast_tanh(e4.y + sA[bi][k + 1]) * sV[k + 1];
        s += fast_tanh(e4.z + sA[bi][k + 2]) * sV[k + 2];
        s += fast_tanh(e4.w + sA[bi][k + 3]) * sV[k + 3];
      }
      sS[bi][t] = s;
    }
    __syncthreads();

    // (6) softmax over t (write probs), argmax (first max), gather next ptr.
    if (wid < ROWS) {
      const int bi = wid;
      float lv[8];
      float bv = -3.4e38f; int bidx = 0;
#pragma unroll
      for (int i = 0; i < 8; ++i) {
        int t = lane + 32 * i;
        float s = sS[bi][t];
        lv[i] = s;
        if (s > bv) { bv = s; bidx = t; }          // strict '>' keeps first occurrence
      }
#pragma unroll
      for (int off = 16; off > 0; off >>= 1) {
        float ov = __shfl_xor(bv, off, 32);
        int   oi = __shfl_xor(bidx, off, 32);
        if (ov > bv || (ov == bv && oi < bidx)) { bv = ov; bidx = oi; }
      }
      float sum = 0.f; float evv[8];
#pragma unroll
      for (int i = 0; i < 8; ++i) { evv[i] = fast_exp(lv[i] - bv); sum += evv[i]; }
#pragma unroll
      for (int off = 16; off > 0; off >>= 1) sum += __shfl_xor(sum, off, 32);
      float rs = __builtin_amdgcn_rcpf(sum);
      float* orow = out + ((size_t)(b0 + bi) * Tsz + step) * Tsz;
#pragma unroll
      for (int i = 0; i < 8; ++i) orow[lane + 32 * i] = evv[i] * rs;
      if (lane == 0) {
        const float* xr = x + ((size_t)(b0 + bi) * Tsz + bidx) * 2;
        sP[bi][0] = xr[0];
        sP[bi][1] = xr[1];
      }
    }
    __syncthreads();   // next step's z-init reads sP
  }
}

// ---------------------------------------------------------------------------
extern "C" void kernel_launch(void* const* d_in, const int* in_sizes, int n_in,
                              void* d_out, int out_size, void* d_ws, size_t ws_size,
                              hipStream_t stream) {
  (void)in_sizes; (void)n_in; (void)out_size; (void)ws_size;
  const float* x   = (const float*)d_in[0];
  const float* enc = (const float*)d_in[1];
  const float* h0  = (const float*)d_in[2];
  const float* c0  = (const float*)d_in[3];
  const float* W1  = (const float*)d_in[4];
  const float* W2  = (const float*)d_in[5];
  const float* V   = (const float*)d_in[6];
  const float* Wk  = (const float*)d_in[7];
  const float* Wr  = (const float*)d_in[8];
  const float* b   = (const float*)d_in[9];
  float* out = (float*)d_out;

  // workspace: enc_proj [64*256*128] f32 (8 MB) + Wr bf16 (128 KB) + W2 bf16 (32 KB)
  float*  ep   = (float*)d_ws;
  __bf16* wrbf = (__bf16*)((char*)d_ws + (size_t)Bsz * Tsz * Hsz * sizeof(float));
  __bf16* w2bf = wrbf + (size_t)Hsz * H4sz;

  enc_proj_gemm<<<1024, 256, 0, stream>>>(enc, W1, ep);
  weights_to_bf16<<<(Hsz * H4sz + Hsz * Hsz) / 256, 256, 0, stream>>>(Wr, W2, wrbf, w2bf);
  ptrnet_decoder<<<Bsz / ROWS, 256, 0, stream>>>(x, ep, wrbf, w2bf, h0, c0, V, Wk, b, out);
}
